// ImplicitRNNCell_83975200572027
// MI455X (gfx1250) — matrix operand order repaired and checked by reading, not compiled, not run on hardware
//
#include <hip/hip_runtime.h>
#include <hip/hip_bf16.h>

// ---- problem constants (match reference) ----
#define B_SZ   512
#define SEQL   64
#define INDIM  128
#define HIDD   256
#define NIMP   512
#define PDIM   384      // INDIM + HIDD
#define NITER  16

// ---- tiling ----
// BTILE=32: two N=16 column groups per workgroup -> each streamed A fragment
// feeds 2 WMMAs (halves L2 traffic vs BTILE=16). 512/32 = 16 workgroups.
#define BTILE  32
#define NTHR   256      // 8 waves; 2 waves/SIMD -> generous per-wave VGPR cap
#define USTR   400      // ushorts per U column  (384 + pad; 800B, 800%256==32 -> bank spread)
#define XSTR   528      // ushorts per X column  (512 + pad; 1056B, 1056%256==32)

#define LDS_U_OFF   0
#define LDS_X0_OFF  (BTILE * USTR)                  // after U
#define LDS_X1_OFF  (LDS_X0_OFF + BTILE * XSTR)
#define LDS_TOTAL   ((LDS_X1_OFF + BTILE * XSTR) * 2)   // bytes

typedef __attribute__((ext_vector_type(16))) __bf16 v16bf;
typedef __attribute__((ext_vector_type(8)))  float  v8f;

struct Frag {
    union { v16bf v; uint4 q[2]; unsigned short s[16]; };
};

__device__ __forceinline__ unsigned short f2bf(float f) {
    unsigned int u = __float_as_uint(f);
    unsigned int r = (u + 0x7FFFu + ((u >> 16) & 1u)) >> 16;   // round-to-nearest-even
    return (unsigned short)r;
}

// WMMA bf16: D = A*B + C, fp32 accumulate (v_wmma_f32_16x16x32_bf16)
__device__ __forceinline__ v8f wmma_bf16(const Frag& a, const Frag& b, v8f c) {
    return __builtin_amdgcn_wmma_f32_16x16x32_bf16(
        /*neg_a=*/false, a.v, /*neg_b=*/false, b.v,
        /*c_mod=*/(short)0, c, /*reuse_a=*/false, /*reuse_b=*/false);
}

// Freeze instruction order across pipeline stages: without this the machine
// scheduler sinks prefetch loads to just before their use (round 5 asm:
// s_wait_loadcnt 0x0 before every WMMA pair) and RA coalesces the rotation
// buffers, collapsing the software pipeline.
#define PIPELINE_FENCE() __builtin_amdgcn_sched_barrier(0)

// Opaque zero: an offset the compiler cannot fold away. Adding it to a global
// pointer defeats LICM/register-promotion of whole weight matrices (the spill
// cause in rounds 1-2) while PRESERVING address-space provenance, so streams
// stay global_load_b128 with decoupled LOADcnt waits (flat would couple DScnt).
__device__ __forceinline__ size_t opaque_zero() {
    size_t z = 0;
    asm volatile("" : "+s"(z));
    return z;
}

// Streamed A-operand fragment (global, pre-swizzled): 1KB/fragment, lane owns 32B.
__device__ __forceinline__ Frag ld_afrag(const unsigned short* base, int lane) {
    Frag f;
    const uint4* p = (const uint4*)base + lane * 2;
    f.q[0] = p[0];
    f.q[1] = p[1];
    return f;
}

// B-operand fragment from LDS (column-major bf16 matrix):
// lane l -> column (l&15); h=l>>4: elems 0-7 K=[kb*32+8h,+8), elems 8-15 K=[kb*32+16+8h,+8).
__device__ __forceinline__ Frag ld_bfrag(const unsigned short* lds, int colStride,
                                         int kb, int col, int khalf) {
    Frag f;
    const unsigned short* p = lds + col * colStride + kb * 32 + khalf * 8;
    f.q[0] = *(const uint4*)(p);
    f.q[1] = *(const uint4*)(p + 16);
    return f;
}

// Store a 16x16 f32 accumulator tile as relu()'d bf16 into column-major LDS X buffer.
// C/D layout: VGPR r, lanes 0-15 -> M = r (N = lane), lanes 16-31 -> M = 8+r.
__device__ __forceinline__ void store_x_tile(unsigned short* xbuf, int mb,
                                             int col, int khalf, v8f acc) {
    int rowbase = mb * 16 + khalf * 8;
    unsigned short* p = xbuf + col * XSTR + rowbase;
#pragma unroll
    for (int r = 0; r < 8; r += 2) {
        float a0 = acc[r]     > 0.f ? acc[r]     : 0.f;
        float a1 = acc[r + 1] > 0.f ? acc[r + 1] : 0.f;
        unsigned int pk = (unsigned int)f2bf(a0) | ((unsigned int)f2bf(a1) << 16);
        *(unsigned int*)(p + r) = pk;   // ds_store_b32, rows r,r+1 adjacent
    }
}

// ---- one-time weight swizzle into WMMA A-operand fragment order (bf16) ----
// Fragment (mb,kb) covers rows [mb*16,+16), K [kb*32,+32).  Per ISA 7.12.2 (16-bit A 16x32):
// lane l: m = l&15, khalf = l>>4; VGPR j holds K = (j<4?0:16) + khalf*8 + (j&3)*2 (+1 hi half).
__global__ void swizzle_kernel(const float* __restrict__ src, unsigned short* __restrict__ dst,
                               int nmb, int nkb, int K) {
    int frag = blockIdx.x * (blockDim.x >> 5) + (threadIdx.x >> 5);
    int lane = threadIdx.x & 31;
    if (frag >= nmb * nkb) return;
    int mb = frag / nkb, kb = frag % nkb;
    int m = lane & 15, khalf = lane >> 4;
    unsigned short tmp[16];
#pragma unroll
    for (int e = 0; e < 16; ++e) {
        int j = e >> 1, p = e & 1;
        int k = kb * 32 + ((j < 4) ? 0 : 16) + khalf * 8 + (j & 3) * 2 + p;
        tmp[e] = f2bf(src[(size_t)(mb * 16 + m) * K + k]);
    }
    uint4* o = (uint4*)(dst + (size_t)frag * 512 + lane * 16);
    o[0] = *(uint4*)&tmp[0];
    o[1] = *(uint4*)&tmp[8];
}

// ---- persistent per-batch-slice RNN kernel: 16 blocks x 256 threads (8 waves) ----
__global__ __launch_bounds__(NTHR, 1)
void implicit_rnn_kernel(const float* __restrict__ x,
                         const unsigned short* __restrict__ Asw,   // 32x16 frags
                         const unsigned short* __restrict__ Bsw,   // 32x12 frags
                         const unsigned short* __restrict__ Csw,   // 16x16 frags
                         const unsigned short* __restrict__ Dsw,   // 16x12 frags
                         float* __restrict__ outSeq,               // (B, SEQ, HID)
                         float* __restrict__ outLast) {            // (B, HID)
    extern __shared__ unsigned short smem[];
    unsigned short* ldsU  = smem + LDS_U_OFF;    // u = [x_t ; h], bf16 col-major, 32 cols
    unsigned short* ldsX0 = smem + LDS_X0_OFF;   // X buffer 0
    unsigned short* ldsX1 = smem + LDS_X1_OFF;   // X buffer 1

    const int tid   = threadIdx.x;
    const int lane  = tid & 31;
    const int wave  = tid >> 5;        // 0..7
    const int col   = lane & 15;
    const int khalf = lane >> 4;
    const int colBase = blockIdx.x * BTILE;

    // h0 = 0: zero the U buffer (x part gets overwritten each step)
    for (int i = tid; i < BTILE * USTR; i += NTHR) ldsU[i] = 0;
    __syncthreads();

    for (int t = 0; t < SEQL; ++t) {
        // ---- Phase A: load x_t (f32 global) -> bf16 U[0:128], 32 columns ----
        {
            int c = tid >> 3, s = tid & 7;   // 8 threads/column, 16 elems each
            const float* src = x + ((size_t)(colBase + c) * SEQL + t) * INDIM + s * 16;
            unsigned short tmp[16];
#pragma unroll
            for (int e = 0; e < 16; ++e) tmp[e] = f2bf(src[e]);
            *(uint4*)&ldsU[c * USTR + s * 16]     = *(uint4*)&tmp[0];
            *(uint4*)&ldsU[c * USTR + s * 16 + 8] = *(uint4*)&tmp[8];
        }
        __syncthreads();

        // ---- Phase B: BU = Bmat @ u (registers), X1 = relu(BU) ----
        v8f bu[4][2];
#pragma unroll
        for (int tt = 0; tt < 4; ++tt) { bu[tt][0] = {}; bu[tt][1] = {}; }
        {
            const unsigned short* Bp = Bsw + opaque_zero();
            Frag wbuf[3][4];   // weight stream, 2 stages ahead
            Frag bbuf[2][2];   // U operand stream, 1 stage ahead
#pragma unroll
            for (int pk = 0; pk < 2; ++pk)
#pragma unroll
                for (int tt = 0; tt < 4; ++tt)
                    wbuf[pk][tt] = ld_afrag(
                        Bp + (size_t)((wave * 4 + tt) * 12 + pk) * 512, lane);
            bbuf[0][0] = ld_bfrag(ldsU, USTR, 0, col,      khalf);
            bbuf[0][1] = ld_bfrag(ldsU, USTR, 0, col + 16, khalf);
            PIPELINE_FENCE();
#pragma unroll
            for (int kb = 0; kb < 12; ++kb) {       // P = 384 = 12 * 32
                if (kb + 2 < 12) {
#pragma unroll
                    for (int tt = 0; tt < 4; ++tt)
                        wbuf[(kb + 2) % 3][tt] = ld_afrag(
                            Bp + (size_t)((wave * 4 + tt) * 12 + kb + 2) * 512, lane);
                }
                if (kb + 1 < 12) {
                    bbuf[(kb + 1) & 1][0] = ld_bfrag(ldsU, USTR, kb + 1, col,      khalf);
                    bbuf[(kb + 1) & 1][1] = ld_bfrag(ldsU, USTR, kb + 1, col + 16, khalf);
                }
#pragma unroll
                for (int tt = 0; tt < 4; ++tt) {    // 32 row-blocks / 8 waves
                    bu[tt][0] = wmma_bf16(wbuf[kb % 3][tt], bbuf[kb & 1][0], bu[tt][0]);
                    bu[tt][1] = wmma_bf16(wbuf[kb % 3][tt], bbuf[kb & 1][1], bu[tt][1]);
                }
                PIPELINE_FENCE();
            }
        }
#pragma unroll
        for (int tt = 0; tt < 4; ++tt) {
            store_x_tile(ldsX0, wave * 4 + tt, col,      khalf, bu[tt][0]);  // X1 = relu(bu)
            store_x_tile(ldsX0, wave * 4 + tt, col + 16, khalf, bu[tt][1]);
        }
        __syncthreads();

        // ---- Phase C: 15 more fixed-point iterations X <- relu(A X + bu) ----
        int cur = 0;
        for (int it = 1; it < NITER; ++it) {
            const unsigned short* Ap = Asw + opaque_zero();   // defeat LICM, keep global AS
            unsigned short* xc = cur ? ldsX1 : ldsX0;
            unsigned short* xn = cur ? ldsX0 : ldsX1;
            v8f acc[4][2];
#pragma unroll
            for (int tt = 0; tt < 4; ++tt) { acc[tt][0] = bu[tt][0]; acc[tt][1] = bu[tt][1]; }

            Frag abuf[3][4];   // A stream, 2 stages ahead
            Frag bbuf[2][2];   // X operand stream, 1 stage ahead
#pragma unroll
            for (int pk = 0; pk < 2; ++pk)
#pragma unroll
                for (int tt = 0; tt < 4; ++tt)
                    abuf[pk][tt] = ld_afrag(
                        Ap + (size_t)((wave * 4 + tt) * 16 + pk) * 512, lane);
            bbuf[0][0] = ld_bfrag(xc, XSTR, 0, col,      khalf);
            bbuf[0][1] = ld_bfrag(xc, XSTR, 0, col + 16, khalf);
            PIPELINE_FENCE();
#pragma unroll
            for (int kb = 0; kb < 16; ++kb) {       // n = 512 = 16 * 32
                if (kb + 2 < 16) {
#pragma unroll
                    for (int tt = 0; tt < 4; ++tt)
                        abuf[(kb + 2) % 3][tt] = ld_afrag(
                            Ap + (size_t)((wave * 4 + tt) * 16 + kb + 2) * 512, lane);
                }
                if (kb + 1 < 16) {
                    bbuf[(kb + 1) & 1][0] = ld_bfrag(xc, XSTR, kb + 1, col,      khalf);
                    bbuf[(kb + 1) & 1][1] = ld_bfrag(xc, XSTR, kb + 1, col + 16, khalf);
                }
#pragma unroll
                for (int tt = 0; tt < 4; ++tt) {
                    acc[tt][0] = wmma_bf16(abuf[kb % 3][tt], bbuf[kb & 1][0], acc[tt][0]);
                    acc[tt][1] = wmma_bf16(abuf[kb % 3][tt], bbuf[kb & 1][1], acc[tt][1]);
                }
                PIPELINE_FENCE();
            }
#pragma unroll
            for (int tt = 0; tt < 4; ++tt) {
                store_x_tile(xn, wave * 4 + tt, col,      khalf, acc[tt][0]);
                store_x_tile(xn, wave * 4 + tt, col + 16, khalf, acc[tt][1]);
            }
            __syncthreads();
            cur ^= 1;
        }

        // ---- Phase D: h = C @ X + D @ u ; write outputs, feed h back into U ----
        {
            const unsigned short* Cp = Csw + opaque_zero();
            const unsigned short* Dp = Dsw + opaque_zero();
            unsigned short* xc = cur ? ldsX1 : ldsX0;
            v8f h[2][2];
#pragma unroll
            for (int hh = 0; hh < 2; ++hh) { h[hh][0] = {}; h[hh][1] = {}; }

            // Unified 28-stage stream (16 C frags then 12 D frags), pipelined.
            auto haddr = [&](int i, int hh) -> const unsigned short* {
                return (i < 16) ? Cp + (size_t)((wave * 2 + hh) * 16 + i) * 512
                                : Dp + (size_t)((wave * 2 + hh) * 12 + (i - 16)) * 512;
            };
            auto hbfrag = [&](int i, int cg) -> Frag {
                return (i < 16) ? ld_bfrag(xc,   XSTR, i,      col + cg * 16, khalf)
                                : ld_bfrag(ldsU, USTR, i - 16, col + cg * 16, khalf);
            };
            Frag dbuf[3][2];
            Frag bbuf[2][2];
#pragma unroll
            for (int pk = 0; pk < 2; ++pk)
#pragma unroll
                for (int hh = 0; hh < 2; ++hh)
                    dbuf[pk][hh] = ld_afrag(haddr(pk, hh), lane);
            bbuf[0][0] = hbfrag(0, 0);
            bbuf[0][1] = hbfrag(0, 1);
            PIPELINE_FENCE();
#pragma unroll
            for (int i = 0; i < 28; ++i) {
                if (i + 2 < 28) {
#pragma unroll
                    for (int hh = 0; hh < 2; ++hh)
                        dbuf[(i + 2) % 3][hh] = ld_afrag(haddr(i + 2, hh), lane);
                }
                if (i + 1 < 28) {
                    bbuf[(i + 1) & 1][0] = hbfrag(i + 1, 0);
                    bbuf[(i + 1) & 1][1] = hbfrag(i + 1, 1);
                }
#pragma unroll
                for (int hh = 0; hh < 2; ++hh) {    // 16 H row-blocks / 8 waves
                    h[hh][0] = wmma_bf16(dbuf[i % 3][hh], bbuf[i & 1][0], h[hh][0]);
                    h[hh][1] = wmma_bf16(dbuf[i % 3][hh], bbuf[i & 1][1], h[hh][1]);
                }
                PIPELINE_FENCE();
            }
#pragma unroll
            for (int hh = 0; hh < 2; ++hh) {
#pragma unroll
                for (int nb = 0; nb < 2; ++nb) {
                    v8f acc = h[hh][nb];
                    int bcol    = colBase + nb * 16 + col;
                    int rowbase = (wave * 2 + hh) * 16 + khalf * 8;
                    float* op = outSeq + ((size_t)bcol * SEQL + t) * HIDD + rowbase;
#pragma unroll
                    for (int r = 0; r < 8; ++r) op[r] = acc[r];
                    if (t == SEQL - 1) {
                        float* lp = outLast + (size_t)bcol * HIDD + rowbase;
#pragma unroll
                        for (int r = 0; r < 8; ++r) lp[r] = acc[r];
                    }
                    // h -> bf16 into U[128:384] for next step
                    unsigned short* up = ldsU + (nb * 16 + col) * USTR + INDIM + rowbase;
#pragma unroll
                    for (int r = 0; r < 8; r += 2) {
                        unsigned int pk = (unsigned int)f2bf(acc[r]) |
                                          ((unsigned int)f2bf(acc[r + 1]) << 16);
                        *(unsigned int*)(up + r) = pk;
                    }
                }
            }
        }
        __syncthreads();
    }
}

extern "C" void kernel_launch(void* const* d_in, const int* in_sizes, int n_in,
                              void* d_out, int out_size, void* d_ws, size_t ws_size,
                              hipStream_t stream) {
    const float* x  = (const float*)d_in[0];   // (512, 64, 128)
    const float* A  = (const float*)d_in[1];   // (512, 512)
    const float* Bm = (const float*)d_in[2];   // (512, 384)
    const float* C  = (const float*)d_in[3];   // (256, 512)
    const float* D  = (const float*)d_in[4];   // (256, 384)

    unsigned short* Asw = (unsigned short*)d_ws;               // 512*512 bf16 = 512KB
    unsigned short* Bsw = Asw + (size_t)512 * 512;             // 512*384 bf16 = 384KB
    unsigned short* Csw = Bsw + (size_t)512 * 384;             // 256*512 bf16 = 256KB
    unsigned short* Dsw = Csw + (size_t)256 * 512;             // 256*384 bf16 = 192KB

    float* outSeq  = (float*)d_out;
    float* outLast = outSeq + (size_t)B_SZ * SEQL * HIDD;

    // One-time (per launch) weight swizzles into WMMA fragment order, f32 -> bf16.
    swizzle_kernel<<<64, 256, 0, stream>>>(A,  Asw, 32, 16, 512);
    swizzle_kernel<<<48, 256, 0, stream>>>(Bm, Bsw, 32, 12, 384);
    swizzle_kernel<<<32, 256, 0, stream>>>(C,  Csw, 16, 16, 512);
    swizzle_kernel<<<24, 256, 0, stream>>>(D,  Dsw, 16, 12, 384);

    // Persistent batch-sliced RNN: 16 independent workgroups (batch slices of 32),
    // ~93KB dynamic LDS each (U + double-buffered X).
    implicit_rnn_kernel<<<B_SZ / BTILE, NTHR, LDS_TOTAL, stream>>>(
        x, Asw, Bsw, Csw, Dsw, outSeq, outLast);
}
